// MultiHeadAttention_1571958031018
// MI455X (gfx1250) — matrix operand order repaired and checked by reading, not compiled, not run on hardware
//
#include <hip/hip_runtime.h>
#include <hip/hip_bf16.h>

typedef __attribute__((ext_vector_type(16))) _Float16 v16h;
typedef __attribute__((ext_vector_type(8)))  _Float16 v8h;
typedef __attribute__((ext_vector_type(8)))  float    v8f;

#define DMODEL 512
#define SEQ    2048
#define BATCH  4
#define HEADS  8
#define DK     64
#define TOKENS (BATCH*SEQ)   /* 8192 */

// ---------------- fragment helpers (CDNA5 WMMA 16x16x32 f16 layouts) ----------

static __device__ __forceinline__ v8h ld8(const _Float16* p) {
  return *reinterpret_cast<const v8h*>(p);
}
static __device__ __forceinline__ v16h cat16(v8h lo, v8h hi) {
  v16h r;
#pragma unroll
  for (int i = 0; i < 8; ++i) { r[i] = lo[i]; r[i + 8] = hi[i]; }
  return r;
}

// A fragment 16x32 f16: lane m=lane&15, kg=lane>>4.
// Lane holds K=[k+kg*16,+16) contiguous (same k order as B fragment)
static __device__ __forceinline__ v16h load_afrag(const _Float16* base, int ld,
                                                  int k, int m, int kg) {
  const _Float16* p = base + (size_t)m * ld + k + kg * 16;
  return cat16(ld8(p), ld8(p + 8));
}

// B fragment 32x16 f16: lane n=lane&15, kg=lane>>4.
// Lane holds K=[k+kg*16,+16) contiguous of column n == row n of transposed operand.
static __device__ __forceinline__ v16h load_bfrag(const _Float16* base, int ld,
                                                  int k, int n, int kg) {
  const _Float16* p = base + (size_t)n * ld + k + kg * 16;
  return cat16(ld8(p), ld8(p + 8));
}

static __device__ __forceinline__ v8f wmma_f16(v16h a, v16h b, v8f c) {
  return __builtin_amdgcn_wmma_f32_16x16x32_f16(false, a, false, b,
                                                (short)0, c, false, false);
}

static __device__ __forceinline__ float red_max16(float t) {
#pragma unroll
  for (int m = 1; m <= 8; m <<= 1) t = fmaxf(t, __shfl_xor(t, m, 32));
  return t;
}
static __device__ __forceinline__ float red_sum16(float t) {
#pragma unroll
  for (int m = 1; m <= 8; m <<= 1) t += __shfl_xor(t, m, 32);
  return t;
}

// Async 16B copy global -> LDS (CDNA5 ASYNCcnt path).
// lds_byte_off is relative to the workgroup LDS base (single __shared__ array
// placed at offset 0).
static __device__ __forceinline__ void async_cp16(unsigned lds_byte_off,
                                                  const _Float16* gsrc) {
  asm volatile("global_load_async_to_lds_b128 %0, %1, off"
               :: "v"(lds_byte_off), "v"((unsigned long long)(uintptr_t)gsrc)
               : "memory");
}
static __device__ __forceinline__ void wait_async0() {
  asm volatile("s_wait_asynccnt 0" ::: "memory");
}

// ---------------- precision / layout conversion -------------------------------

// 8 elements per thread: 2x float4 load -> v8h store (pure HBM bandwidth pass)
__global__ __launch_bounds__(256) void mha_cvt_x(const float* __restrict__ x,
                                                 _Float16* __restrict__ x16) {
  const int i = (blockIdx.x * 256 + threadIdx.x) * 8;
  const float4 f0 = *reinterpret_cast<const float4*>(x + i);
  const float4 f1 = *reinterpret_cast<const float4*>(x + i + 4);
  v8h o;
  o[0] = (_Float16)f0.x; o[1] = (_Float16)f0.y;
  o[2] = (_Float16)f0.z; o[3] = (_Float16)f0.w;
  o[4] = (_Float16)f1.x; o[5] = (_Float16)f1.y;
  o[6] = (_Float16)f1.z; o[7] = (_Float16)f1.w;
  *reinterpret_cast<v8h*>(x16 + i) = o;
}

// Wt[n*512 + k] = (f16) W[k*512 + n] ; blockIdx.y selects which weight matrix
__global__ __launch_bounds__(256) void mha_transpose_w(
    const float* __restrict__ Wq, const float* __restrict__ Wk,
    const float* __restrict__ Wv, const float* __restrict__ Wo,
    _Float16* __restrict__ Tq, _Float16* __restrict__ Tk,
    _Float16* __restrict__ Tv, _Float16* __restrict__ To) {
  const int sel = blockIdx.y;
  const float* W = (sel == 0) ? Wq : (sel == 1) ? Wk : (sel == 2) ? Wv : Wo;
  _Float16*   Wt = (sel == 0) ? Tq : (sel == 1) ? Tk : (sel == 2) ? Tv : To;
  const int i = blockIdx.x * 256 + threadIdx.x;   // i = k*512 + n
  const int k = i >> 9, n = i & 511;
  Wt[(size_t)n * DMODEL + k] = (_Float16)W[i];
}

// ---------------- GEMM core: 16x64 tile, unroll-by-2 ping-pong ----------------
// Two fragment buffer sets alternate between "being consumed by WMMA" and
// "being filled by loads" -> no register copies, loads always ~1 stage ahead.

#define GEMM_BODY(Aptr, W0ptr)                                                  \
  v8f acc[4] = {};                                                              \
  v16h a0, a1, b0[4], b1[4];                                                    \
  a0 = load_afrag((Aptr), DMODEL, 0, m16, kg);                                  \
  _Pragma("unroll")                                                             \
  for (int c = 0; c < 4; ++c)                                                   \
    b0[c] = load_bfrag((W0ptr) + (size_t)(c * 16) * DMODEL, DMODEL, 0,          \
                       m16, kg);                                                \
  for (int k = 0; k < DMODEL; k += 64) {                                        \
    a1 = load_afrag((Aptr), DMODEL, k + 32, m16, kg);                           \
    _Pragma("unroll")                                                           \
    for (int c = 0; c < 4; ++c)                                                 \
      b1[c] = load_bfrag((W0ptr) + (size_t)(c * 16) * DMODEL, DMODEL, k + 32,   \
                         m16, kg);                                              \
    _Pragma("unroll")                                                           \
    for (int c = 0; c < 4; ++c) acc[c] = wmma_f16(a0, b0[c], acc[c]);           \
    if (k + 64 < DMODEL) {                                                      \
      a0 = load_afrag((Aptr), DMODEL, k + 64, m16, kg);                         \
      _Pragma("unroll")                                                         \
      for (int c = 0; c < 4; ++c)                                               \
        b0[c] = load_bfrag((W0ptr) + (size_t)(c * 16) * DMODEL, DMODEL, k + 64, \
                           m16, kg);                                            \
    }                                                                           \
    _Pragma("unroll")                                                           \
    for (int c = 0; c < 4; ++c) acc[c] = wmma_f16(a1, b1[c], acc[c]);           \
  }

// ---------------- fused QKV projection ----------------------------------------
// Q,K stored [token][512] f16 ; V stored transposed per head [(b*8+h)*64+d][s].

__global__ __launch_bounds__(128) void mha_gemm_qkv(
    const _Float16* __restrict__ x16,
    const _Float16* __restrict__ wqt, const _Float16* __restrict__ wkt,
    const _Float16* __restrict__ wvt,
    const float* __restrict__ bq, const float* __restrict__ bk,
    const float* __restrict__ bv,
    _Float16* __restrict__ q16, _Float16* __restrict__ k16,
    _Float16* __restrict__ vt16) {
  const int lane = threadIdx.x & 31, wid = threadIdx.x >> 5;
  const int m16 = lane & 15, kg = lane >> 4;
  const int sel = blockIdx.y;
  const _Float16* Wt  = (sel == 0) ? wqt : (sel == 1) ? wkt : wvt;
  const float*    bia = (sel == 0) ? bq  : (sel == 1) ? bk  : bv;

  const int wtile = blockIdx.x * 4 + wid;      // 0..4095
  const int tm = wtile >> 3;                   // 512 row tiles of 16 tokens
  const int tn = wtile & 7;                    // 8 col blocks of 64 (== head)

  const _Float16* A = x16 + (size_t)tm * 16 * DMODEL;
  const _Float16* W0 = Wt + (size_t)(tn * 64) * DMODEL;

  GEMM_BODY(A, W0)

#pragma unroll
  for (int c = 0; c < 4; ++c) {
    const float bb = bia[tn * 64 + c * 16 + m16];
#pragma unroll
    for (int v = 0; v < 8; ++v) {
      const float val = acc[c][v] + bb;
      const int row = tm * 16 + v + 8 * kg;    // global token
      if (sel == 2) {                          // V: transposed per-head store
        const int b_ = row >> 11, s_ = row & (SEQ - 1);
        vt16[((size_t)(b_ * HEADS + tn) * DK + c * 16 + m16) * SEQ + s_] =
            (_Float16)val;
      } else {
        _Float16* o = sel ? k16 : q16;
        o[(size_t)row * DMODEL + tn * 64 + c * 16 + m16] = (_Float16)val;
      }
    }
  }
}

// ---------------- flash attention ---------------------------------------------
// Block = 4 waves sharing one (b,h); each wave owns one 16-row q-tile.
// K/V key-blocks (32 keys) staged into LDS with async copies, double-buffered:
// copy of block j+1 overlaps WMMA+softmax of block j.
//
// LDS map (halves), single array => workgroup LDS offset 0:
//   [    0, 4096)  K stages: 2 x (32 keys x 64 d)
//   [ 4096, 8192)  V stages: 2 x (64 d x 32 keys)
//   [ 8192,10240)  P slabs:  4 waves x (16 x 32)

#define LDS_K(st)   ((st) * 2048)
#define LDS_V(st)   (4096 + (st) * 2048)
#define LDS_P(w)    (8192 + (w) * 512)

static __device__ __forceinline__ void stage_kv(const _Float16* Kb,
                                                const _Float16* Vb,
                                                int jb, int st, int tid) {
  const int c0 = tid * 2;
#pragma unroll
  for (int i = 0; i < 2; ++i) {            // K: 256 b128 chunks (32x64 f16)
    const int c = c0 + i;
    const int row = c >> 3, off = (c & 7) * 8;
    async_cp16((unsigned)((LDS_K(st) + row * 64 + off) * 2),
               Kb + (size_t)(jb + row) * DMODEL + off);
  }
#pragma unroll
  for (int i = 0; i < 2; ++i) {            // V: 256 b128 chunks (64x32 f16)
    const int c = c0 + i;
    const int row = c >> 2, off = (c & 3) * 8;
    async_cp16((unsigned)((LDS_V(st) + row * 32 + off) * 2),
               Vb + (size_t)row * SEQ + jb + off);
  }
}

__global__ __launch_bounds__(128) void mha_attn(
    const _Float16* __restrict__ Q, const _Float16* __restrict__ K,
    const _Float16* __restrict__ Vt, _Float16* __restrict__ Ctx) {
  __shared__ __align__(16) _Float16 smem[10240];
  const int tid = threadIdx.x;
  const int lane = tid & 31, wid = tid >> 5;
  const int m16 = lane & 15, kg = lane >> 4;

  const int bh = blockIdx.x >> 5;          // 0..31 : (b,h)
  const int qg = blockIdx.x & 31;          // 32 q-groups of 4 tiles
  const int b  = bh >> 3, h = bh & 7;
  const int qt = qg * 4 + wid;             // this wave's q-tile (0..127)

  const _Float16* Qb = Q + ((size_t)(b * SEQ) + qt * 16) * DMODEL + h * DK;
  const _Float16* Kb = K + (size_t)(b * SEQ) * DMODEL + h * DK;
  const _Float16* Vb = Vt + (size_t)((b * HEADS + h) * DK) * SEQ;
  _Float16* lp = smem + LDS_P(wid);

  // Q resident: 16x64 as two 16x32 A-fragments
  const v16h qa0 = load_afrag(Qb, DMODEL, 0,  m16, kg);
  const v16h qa1 = load_afrag(Qb, DMODEL, 32, m16, kg);

  v8f acc[4] = {};
  float mrow[8], lrow[8];
#pragma unroll
  for (int v = 0; v < 8; ++v) { mrow[v] = -1e30f; lrow[v] = 0.f; }

  stage_kv(Kb, Vb, 0, 0, tid);             // prologue copy
  int st = 0;

  for (int jb = 0; jb < SEQ; jb += 32) {
    wait_async0();                         // this wave's share of stage `st`
    __syncthreads();                       // all waves' shares landed
    if (jb + 32 < SEQ) stage_kv(Kb, Vb, jb + 32, st ^ 1, tid);

    const _Float16* Ks = smem + LDS_K(st); // [key][64]
    const _Float16* Vs = smem + LDS_V(st); // [d][32]

    // ---- scores: two 16x16 tiles over d_k=64 (2 WMMA each), K from LDS ----
    v8f s0 = {}, s1 = {};
    {
      const _Float16* kr0 = Ks + (m16) * 64;        // keys jb..jb+15
      const _Float16* kr1 = Ks + (16 + m16) * 64;   // keys jb+16..+31
      v16h kb;
      kb = cat16(ld8(kr0 + kg * 16),      ld8(kr0 + kg * 16 + 8));
      s0 = wmma_f16(qa0, kb, s0);
      kb = cat16(ld8(kr0 + 32 + kg * 16), ld8(kr0 + 32 + kg * 16 + 8));
      s0 = wmma_f16(qa1, kb, s0);
      kb = cat16(ld8(kr1 + kg * 16),      ld8(kr1 + kg * 16 + 8));
      s1 = wmma_f16(qa0, kb, s1);
      kb = cat16(ld8(kr1 + 32 + kg * 16), ld8(kr1 + 32 + kg * 16 + 8));
      s1 = wmma_f16(qa1, kb, s1);
    }
    s0 = s0 * 0.125f;                      // 1/sqrt(64)
    s1 = s1 * 0.125f;

    // ---- online softmax (row m = v + 8*kg lives across 16 lanes) ----
#pragma unroll
    for (int v = 0; v < 8; ++v) {
      const float rmax = red_max16(fmaxf(s0[v], s1[v]));
      const float mnew = fmaxf(mrow[v], rmax);
      const float alpha = __expf(mrow[v] - mnew);
      const float p0 = __expf(s0[v] - mnew);
      const float p1 = __expf(s1[v] - mnew);
      lrow[v] = lrow[v] * alpha + red_sum16(p0 + p1);
      mrow[v] = mnew;
      acc[0][v] *= alpha; acc[1][v] *= alpha;
      acc[2][v] *= alpha; acc[3][v] *= alpha;
      const int row = v + 8 * kg;
      lp[row * 32 + m16]      = (_Float16)p0;
      lp[row * 32 + 16 + m16] = (_Float16)p1;
    }
    // Re-shape P (C-layout) -> A-layout via per-wave LDS slab; same-wave DS
    // ops are in-order, wait for visibility before re-reading.
    asm volatile("s_wait_dscnt 0" ::: "memory");
    const v16h pa = load_afrag(lp, 32, 0, m16, kg);

    // ---- ctx += P(16x32) x V(32x64): 4 WMMA, V from LDS ----
#pragma unroll
    for (int c = 0; c < 4; ++c) {
      const _Float16* vr = Vs + (size_t)(c * 16 + m16) * 32 + kg * 16;
      v16h vb = cat16(ld8(vr), ld8(vr + 8));
      acc[c] = wmma_f16(pa, vb, acc[c]);
    }
    st ^= 1;
  }

  // ---- normalize + store ctx (f16, [token][h*64+d]) ----
  float inv[8];
#pragma unroll
  for (int v = 0; v < 8; ++v) inv[v] = 1.0f / lrow[v];
#pragma unroll
  for (int c = 0; c < 4; ++c) {
#pragma unroll
    for (int v = 0; v < 8; ++v) {
      const int row = qt * 16 + v + 8 * kg;
      Ctx[((size_t)(b * SEQ) + row) * DMODEL + h * DK + c * 16 + m16] =
          (_Float16)(acc[c][v] * inv[v]);
    }
  }
}

// ---------------- output projection: out = ctx @ Wo + bo (f32 out) ------------

__global__ __launch_bounds__(128) void mha_gemm_out(
    const _Float16* __restrict__ c16, const _Float16* __restrict__ wot,
    const float* __restrict__ bo, float* __restrict__ out) {
  const int lane = threadIdx.x & 31, wid = threadIdx.x >> 5;
  const int m16 = lane & 15, kg = lane >> 4;
  const int wtile = blockIdx.x * 4 + wid;
  const int tm = wtile >> 3;
  const int tn = wtile & 7;

  const _Float16* A = c16 + (size_t)tm * 16 * DMODEL;
  const _Float16* W0 = wot + (size_t)(tn * 64) * DMODEL;

  GEMM_BODY(A, W0)

#pragma unroll
  for (int c = 0; c < 4; ++c) {
    const float bb = bo[tn * 64 + c * 16 + m16];
#pragma unroll
    for (int v = 0; v < 8; ++v) {
      const int row = tm * 16 + v + 8 * kg;
      out[(size_t)row * DMODEL + tn * 64 + c * 16 + m16] = acc[c][v] + bb;
    }
  }
}

// ---------------- host-side launch --------------------------------------------

extern "C" void kernel_launch(void* const* d_in, const int* in_sizes, int n_in,
                              void* d_out, int out_size, void* d_ws, size_t ws_size,
                              hipStream_t stream) {
  const float* x  = (const float*)d_in[0];
  const float* Wq = (const float*)d_in[1];
  const float* bq = (const float*)d_in[2];
  const float* Wk = (const float*)d_in[3];
  const float* bk = (const float*)d_in[4];
  const float* Wv = (const float*)d_in[5];
  const float* bv = (const float*)d_in[6];
  const float* Wo = (const float*)d_in[7];
  const float* bo = (const float*)d_in[8];
  float* out = (float*)d_out;

  char* ws = (char*)d_ws;
  _Float16* x16  = (_Float16*)(ws + 0);                               // 8 MiB
  _Float16* wqt  = (_Float16*)(ws + ((size_t)8  << 20));              // 512 KiB
  _Float16* wkt  = (_Float16*)(ws + ((size_t)8  << 20) + (512u << 10));
  _Float16* wvt  = (_Float16*)(ws + ((size_t)8  << 20) + (1024u << 10));
  _Float16* wot  = (_Float16*)(ws + ((size_t)8  << 20) + (1536u << 10));
  _Float16* q16  = (_Float16*)(ws + ((size_t)10 << 20));              // 8 MiB
  _Float16* k16  = (_Float16*)(ws + ((size_t)18 << 20));              // 8 MiB
  _Float16* vt16 = (_Float16*)(ws + ((size_t)26 << 20));              // 8 MiB
  _Float16* c16  = (_Float16*)(ws + ((size_t)34 << 20));              // 8 MiB

  mha_cvt_x<<<(TOKENS * DMODEL) / (256 * 8), 256, 0, stream>>>(x, x16);
  mha_transpose_w<<<dim3((DMODEL * DMODEL) / 256, 4), 256, 0, stream>>>(
      Wq, Wk, Wv, Wo, wqt, wkt, wvt, wot);

  // (TOKENS/16)*(DMODEL/64) = 4096 wave tiles, 4 waves per block, y = Q/K/V
  mha_gemm_qkv<<<dim3(1024, 3), 128, 0, stream>>>(x16, wqt, wkt, wvt,
                                                  bq, bk, bv, q16, k16, vt16);

  // 32 (b,h) pairs x 32 q-groups; block = 4 q-tiles sharing staged K/V
  mha_attn<<<1024, 128, 0, stream>>>(q16, k16, vt16, c16);

  mha_gemm_out<<<1024, 128, 0, stream>>>(c16, wot, bo, out);
}
